// MoELayer_71047349011193
// MI455X (gfx1250) — matrix operand order repaired and checked by reading, not compiled, not run on hardware
//
#include <hip/hip_runtime.h>

// ---------------------------------------------------------------------------
// MoE (top-2 of 8 experts) for gfx1250: bf16 WMMA gathered expert GEMM with
// async global->LDS staging (ASYNCcnt double buffering), 128x128 WG tiles.
// ---------------------------------------------------------------------------

typedef __attribute__((ext_vector_type(16))) __bf16 v16bf;
typedef __attribute__((ext_vector_type(8)))  float  v8f;

struct alignas(16) U128 { unsigned int u[4]; };
struct alignas(16) U256 { U128 lo, hi; };

__device__ __forceinline__ unsigned short f2bf(float f) {
    unsigned int u = __builtin_bit_cast(unsigned int, f);
    u += 0x7FFFu + ((u >> 16) & 1u);   // round-to-nearest-even
    return (unsigned short)(u >> 16);
}

// ---- 1. zero output + expert counters -------------------------------------
__global__ void k_zero(float* __restrict__ out, long n4, int* __restrict__ counts) {
    long i = (long)blockIdx.x * blockDim.x + threadIdx.x;
    if (i < n4) reinterpret_cast<float4*>(out)[i] = make_float4(0.f, 0.f, 0.f, 0.f);
    if (blockIdx.x == 0 && threadIdx.x < 8) counts[threadIdx.x] = 0;
}

// ---- 2. convert activations f32 -> bf16 (row-major [N,D]) -----------------
__global__ void k_cvt_x(const float* __restrict__ x, unsigned short* __restrict__ xh, long n4) {
    long i = (long)blockIdx.x * blockDim.x + threadIdx.x;
    if (i >= n4) return;
    float4 v = reinterpret_cast<const float4*>(x)[i];
    ushort4 o;
    o.x = f2bf(v.x); o.y = f2bf(v.y); o.z = f2bf(v.z); o.w = f2bf(v.w);
    reinterpret_cast<ushort4*>(xh)[i] = o;
}

// ---- 3. transpose+convert expert_w [E,D,H] f32 -> wt [E,H,D] bf16 ---------
__global__ void k_cvt_w(const float* __restrict__ ew, unsigned short* __restrict__ wt,
                        int D, int H) {
    __shared__ unsigned short t[32][33];
    const int e  = blockIdx.z;
    const int h0 = blockIdx.x * 32, k0 = blockIdx.y * 32;
    const int tx = threadIdx.x, ty = threadIdx.y;        // 32 x 8
    const float* src = ew + (size_t)e * D * H;
    unsigned short* dst = wt + (size_t)e * H * D;
    #pragma unroll
    for (int i = 0; i < 32; i += 8)
        t[ty + i][tx] = f2bf(src[(size_t)(k0 + ty + i) * H + h0 + tx]);
    __syncthreads();
    #pragma unroll
    for (int i = 0; i < 32; i += 8)
        dst[(size_t)(h0 + ty + i) * D + k0 + tx] = t[tx][ty + i];
}

// ---- 4. router: logits, top-2, softmax, scatter into per-expert lists -----
__global__ void k_router(const float* __restrict__ x, const float* __restrict__ rw,
                         const float* __restrict__ rb, int* __restrict__ counts,
                         int* __restrict__ idxl, float* __restrict__ wgtl,
                         int N, int D) {
    __shared__ float lg[32][8];
    const int tid = threadIdx.x;               // 256 threads: 32 tokens x 8 experts
    const int tok = blockIdx.x * 32 + (tid >> 3);
    const int e   = tid & 7;
    const float4* xr = reinterpret_cast<const float4*>(x + (size_t)tok * D);
    const float4* wr = reinterpret_cast<const float4*>(rw + (size_t)e * D);
    float acc = 0.f;
    for (int d = 0; d < D / 4; ++d) {
        float4 a = xr[d], b = wr[d];
        acc += a.x * b.x + a.y * b.y + a.z * b.z + a.w * b.w;
    }
    lg[tid >> 3][e] = acc + rb[e];
    __syncthreads();
    if (tid < 32) {
        const int tk = blockIdx.x * 32 + tid;
        float v[8];
        #pragma unroll
        for (int j = 0; j < 8; ++j) v[j] = lg[tid][j];
        int i1 = 0; float v1 = v[0];
        #pragma unroll
        for (int j = 1; j < 8; ++j) if (v[j] > v1) { v1 = v[j]; i1 = j; }  // ties -> lowest idx
        int i2 = -1; float v2 = -3.4e38f;
        #pragma unroll
        for (int j = 0; j < 8; ++j) if (j != i1 && v[j] > v2) { v2 = v[j]; i2 = j; }
        const float w1 = 1.f / (1.f + __expf(v2 - v1));   // softmax over the two
        const float w2 = 1.f / (1.f + __expf(v1 - v2));
        int p1 = atomicAdd(&counts[i1], 1);
        idxl[i1 * N + p1] = tk; wgtl[i1 * N + p1] = w1;
        int p2 = atomicAdd(&counts[i2], 1);
        idxl[i2 * N + p2] = tk; wgtl[i2 * N + p2] = w2;
    }
}

// ---- 5. gathered expert GEMM ----------------------------------------------
// 256-thread WG = 8 waves in a 4(M) x 2(N) grid; WG tile 128x128, wave tile
// 32x64 (2x4 grid of 16x16x32 bf16 WMMAs). A (gathered token rows) and B
// (K-contiguous weights) are staged into LDS with GLOBAL_LOAD_ASYNC_TO_LDS,
// double-buffered on ASYNCcnt. Rows padded to 80B in LDS.
#define LDS_PITCH 80                 // 32 bf16 (64B) padded to 80B
#define LDS_ABUF  (128 * LDS_PITCH)  // 10240 B per A buffer
#define LDS_BUF   (2 * LDS_ABUF)     // A + B per stage buffer
#define ASYNC_B128(ldsoff, gaddr)                                           \
    asm volatile("global_load_async_to_lds_b128 %0, %1, off"                \
                 :: "v"(ldsoff), "v"(gaddr) : "memory")

__global__ __launch_bounds__(256) void k_moe_gemm(
    const unsigned short* __restrict__ xh,   // [N, D]  bf16
    const unsigned short* __restrict__ wt,   // [E, H, D] bf16 (transposed)
    const float* __restrict__ eb,            // [E, H]
    const int* __restrict__ counts,          // [E]
    const int* __restrict__ idxl,            // [E, N]
    const float* __restrict__ wgtl,          // [E, N]
    float* __restrict__ out,                 // [N, H]
    int N, int D, int H) {
    const int e = blockIdx.z;
    const int count = counts[e];
    const int m0 = blockIdx.y * 128;
    if (m0 >= count) return;
    const int n0 = blockIdx.x * 128;
    const int tid  = threadIdx.x;
    const int lane = tid & 31;
    const int wv   = tid >> 5;               // wave 0..7
    const int lo   = lane & 15;
    const int hi   = lane >> 4;
    const int wm   = (wv & 3) * 32;          // wave M offset within WG tile
    const int wn   = (wv >> 2) * 64;         // wave N offset within WG tile

    const int*   idx_e = idxl + (size_t)e * N;
    const float* wgt_e = wgtl + (size_t)e * N;
    const unsigned short* wt_e = wt + (size_t)e * H * D;

    __shared__ __align__(16) unsigned char lds[2 * LDS_BUF];
    const unsigned ldsbase = (unsigned)(size_t)(&lds[0]);  // low 32b = LDS offset

    // Each thread owns 2 copy slots for A and 2 for B (512 x 16B chunks each).
    // slot s: idx = tid + s*256; row = idx>>2 in [0,128), chunk = idx&3.
    unsigned long long gA[2], gB[2];
    unsigned lA[2], lB[2];
    #pragma unroll
    for (int s = 0; s < 2; ++s) {
        int idx = tid + s * 256;
        int row = idx >> 2, ch = idx & 3;
        int gm = m0 + row;
        int tokn = (gm < count) ? idx_e[gm] : 0;          // clamp padded rows
        gA[s] = (unsigned long long)(size_t)(xh + (size_t)tokn * D + ch * 8);
        lA[s] = ldsbase + row * LDS_PITCH + ch * 16;
        gB[s] = (unsigned long long)(size_t)(wt_e + (size_t)(n0 + row) * D + ch * 8);
        lB[s] = ldsbase + LDS_ABUF + row * LDS_PITCH + ch * 16;
    }

    // Prologue: stage k0 = 0 into buffer 0 (4 async instrs per wave).
    #pragma unroll
    for (int s = 0; s < 2; ++s) {
        ASYNC_B128(lA[s], gA[s]);
        ASYNC_B128(lB[s], gB[s]);
    }

    v8f c[2][4] = {};
    int buf = 0;
    for (int k0 = 0; k0 < D; k0 += 32) {
        const int nbuf = buf ^ 1;
        if (k0 + 32 < D) {
            const unsigned boff = (unsigned)nbuf * LDS_BUF;
            const unsigned long long gadv = (unsigned long long)(k0 + 32) * 2; // bytes
            #pragma unroll
            for (int s = 0; s < 2; ++s) {
                ASYNC_B128(lA[s] + boff, gA[s] + gadv);
                ASYNC_B128(lB[s] + boff, gB[s] + gadv);
            }
            asm volatile("s_wait_asynccnt 4" ::: "memory"); // current buffer complete
        } else {
            asm volatile("s_wait_asynccnt 0" ::: "memory");
        }
        __syncthreads();                                    // all waves' copies visible

        const unsigned char* Ab = lds + buf * LDS_BUF;
        const unsigned char* Bb = Ab + LDS_ABUF;
        v16bf a[2], b[4];
        #pragma unroll
        for (int i = 0; i < 2; ++i) {                       // A frag: 32B contiguous per half
            int row = wm + i * 16 + lo;
            U256 u;
            u.lo = *reinterpret_cast<const U128*>(Ab + row * LDS_PITCH + hi * 32);
            u.hi = *reinterpret_cast<const U128*>(Ab + row * LDS_PITCH + hi * 32 + 16);
            a[i] = __builtin_bit_cast(v16bf, u);
        }
        #pragma unroll
        for (int j = 0; j < 4; ++j) {                       // B frag: 32B contiguous per half
            int col = wn + j * 16 + lo;
            U256 u = *reinterpret_cast<const U256*>(Bb + col * LDS_PITCH + hi * 32);
            b[j] = __builtin_bit_cast(v16bf, u);
        }
        #pragma unroll
        for (int i = 0; i < 2; ++i)
            #pragma unroll
            for (int j = 0; j < 4; ++j)
                c[i][j] = __builtin_amdgcn_wmma_f32_16x16x32_bf16(
                    false, a[i], false, b[j], (short)0, c[i][j], false, false);

        __syncthreads();                                    // reads done before overwrite
        buf = nbuf;
    }

    // Epilogue: w * (acc + bias) -> atomic accumulate (exactly 2 adds/element)
    float bias[4];
    #pragma unroll
    for (int j = 0; j < 4; ++j)
        bias[j] = eb[(size_t)e * H + n0 + wn + j * 16 + lo];

    #pragma unroll
    for (int i = 0; i < 2; ++i) {
        #pragma unroll
        for (int v = 0; v < 8; ++v) {
            int gm = m0 + wm + i * 16 + hi * 8 + v;         // C row for this VGPR/lane-half
            if (gm < count) {
                int   tokn = idx_e[gm];
                float w    = wgt_e[gm];
                float* orow = out + (size_t)tokn * H + n0 + wn + lo;
                #pragma unroll
                for (int j = 0; j < 4; ++j) {
                    float val = w * (c[i][j][v] + bias[j]);
                    __hip_atomic_fetch_add(orow + j * 16, val,
                                           __ATOMIC_RELAXED, __HIP_MEMORY_SCOPE_AGENT);
                }
            }
        }
    }
}

// ---------------------------------------------------------------------------
extern "C" void kernel_launch(void* const* d_in, const int* in_sizes, int n_in,
                              void* d_out, int out_size, void* d_ws, size_t ws_size,
                              hipStream_t stream) {
    const float* x  = (const float*)d_in[0];
    const float* rw = (const float*)d_in[1];
    const float* rb = (const float*)d_in[2];
    const float* ew = (const float*)d_in[3];
    const float* eb = (const float*)d_in[4];
    float* out = (float*)d_out;

    const int  E  = in_sizes[2];            // 8
    const int  D  = in_sizes[1] / E;        // 1024
    const long ND = in_sizes[0];            // N*D
    const int  N  = (int)(ND / D);          // 32768
    const int  H  = in_sizes[4] / E;        // 1024

    char* ws = (char*)d_ws;
    size_t off = 0;
    unsigned short* xh = (unsigned short*)(ws + off); off += (size_t)N * D * 2;
    unsigned short* wt = (unsigned short*)(ws + off); off += (size_t)E * H * D * 2;
    int*   counts = (int*)(ws + off);   off += 256;
    int*   idxl   = (int*)(ws + off);   off += (size_t)E * N * sizeof(int);
    float* wgtl   = (float*)(ws + off); off += (size_t)E * N * sizeof(float);
    (void)ws_size; (void)n_in; (void)out_size;

    const long outN4 = (long)N * H / 4;
    k_zero<<<dim3((unsigned)((outN4 + 255) / 256)), 256, 0, stream>>>(out, outN4, counts);

    const long nd4 = ND / 4;
    k_cvt_x<<<dim3((unsigned)((nd4 + 255) / 256)), 256, 0, stream>>>(x, xh, nd4);

    k_cvt_w<<<dim3(H / 32, D / 32, E), dim3(32, 8), 0, stream>>>(ew, wt, D, H);

    k_router<<<dim3(N / 32), 256, 0, stream>>>(x, rw, rb, counts, idxl, wgtl, N, D);

    k_moe_gemm<<<dim3(H / 128, (N + 127) / 128, E), 256, 0, stream>>>(
        xh, wt, eb, counts, idxl, wgtl, out, N, D, H);
}
